// sec_conv_BiLSTM_40742059770622
// MI455X (gfx1250) — matrix operand order repaired and checked by reading, not compiled, not run on hardware
//
#include <hip/hip_runtime.h>
#include <hip/hip_bf16.h>
#include <stdint.h>
#include <stddef.h>

// ---------------------------------------------------------------------------
// Shapes (B=16, W=32, E=1024)
// ---------------------------------------------------------------------------
static constexpr int kB   = 16;
static constexpr int kW   = 32;
static constexpr int kE   = 1024;
static constexpr int kBW  = kB * kW;        // 512 rows
static constexpr int kNG  = 6 * kE;         // 3 live gates x 2 directions = 6144

typedef _Float16 v16h __attribute__((ext_vector_type(16)));
typedef _Float16 v8h  __attribute__((ext_vector_type(8)));
typedef float    v8f  __attribute__((ext_vector_type(8)));

__device__ __forceinline__ float hsig(float x) {
  return fminf(fmaxf(0.2f * x + 0.5f, 0.0f), 1.0f);
}

__device__ __forceinline__ v16h cat16(v8h lo, v8h hi8) {
  return __builtin_shufflevector(lo, hi8, 0,1,2,3,4,5,6,7,8,9,10,11,12,13,14,15);
}

// ---------------------------------------------------------------------------
// Prep kernels: f32 -> f16 conversions / transposes / gate folding
// ---------------------------------------------------------------------------
__global__ __launch_bounds__(256) void prep_x_f16(const float* __restrict__ X,
                                                  _Float16* __restrict__ Xh, int n) {
  int t = blockIdx.x * 256 + threadIdx.x;
  if (t < n) Xh[t] = (_Float16)X[t];
}

// Pack live gate columns of Wf0/Wb0 (E x 4E each) into W1t: (6144 n) x (1024 k) f16,
// n-major rows of K so B-fragments are contiguous.  n = dir*3072 + g*1024 + e,
// g in {i,c,o} -> source gate {0,2,3} (keras order i,f,c,o).
__global__ __launch_bounds__(256) void prep_w1t(const float* __restrict__ Wf0,
                                                const float* __restrict__ Wb0,
                                                _Float16* __restrict__ W1t) {
  int t = blockIdx.x * 256 + threadIdx.x;           // 6144*1024 elements
  if (t >= kNG * kE) return;
  int n = t % kNG;                                   // coalesced source reads
  int k = t / kNG;
  int dir  = n / (3 * kE);
  int rem  = n - dir * 3 * kE;
  int g    = rem >> 10;                              // 0,1,2
  int e    = rem & (kE - 1);
  int gsrc = (g == 0) ? 0 : (g == 1 ? 2 : 3);
  const float* Wsrc = dir ? Wb0 : Wf0;
  W1t[(size_t)n * kE + k] = (_Float16)Wsrc[(size_t)k * 4 * kE + gsrc * kE + e];
}

// Wd1t[e][k] = Wd[k][e]  (first E rows of Wd), f16 transposed.
__global__ __launch_bounds__(256) void prep_wd1t(const float* __restrict__ Wd,
                                                 _Float16* __restrict__ Wd1t) {
  int t = blockIdx.x * 256 + threadIdx.x;            // 1024*1024
  if (t >= kE * kE) return;
  int n = t % kE;
  int k = t / kE;
  Wd1t[(size_t)n * kE + k] = (_Float16)Wd[(size_t)k * kE + n];
}

// Wd2s[g][e] = sum_r Wd[(E + 32g + r)][e]  (folding of the raw-reshape broadcast)
__global__ __launch_bounds__(256) void prep_wd2s(const float* __restrict__ Wd,
                                                 float* __restrict__ Wd2s) {
  int t = blockIdx.x * 256 + threadIdx.x;            // 32*1024
  if (t >= 32 * kE) return;
  int g = t >> 10, e = t & (kE - 1);
  float s = 0.0f;
  #pragma unroll 8
  for (int r = 0; r < 32; ++r) s += Wd[(size_t)(kE + g * 32 + r) * kE + e];
  Wd2s[t] = s;
}

__global__ __launch_bounds__(256) void zero_f32(float* __restrict__ p, int n) {
  int t = blockIdx.x * 256 + threadIdx.x;
  if (t < n) p[t] = 0.0f;
}

__global__ __launch_bounds__(256) void fill_ones(float* __restrict__ p, int n) {
  int t = blockIdx.x * 256 + threadIdx.x;
  if (t < n) p[t] = 1.0f;
}

// ---------------------------------------------------------------------------
// WMMA GEMM, 16x64 strip per wave, software-pipelined K-loop:
//   C(MxN, f32) = A(MxK f16 row-major) x Bt(NxK f16 row-major)
// Fragments for step k+1 are loaded BEFORE the 4 WMMAs of step k, so the
// backend can issue partial s_wait_loadcnt and hide VMEM latency within one
// wave.  A fragment reused across 4 WMMAs -> 1.25 KB operands / 16 KFLOP.
// Fragment layouts per CDNA5 ISA 7.12.2.  Requires M%16==0, N%64==0, K%32==0.
// ---------------------------------------------------------------------------
__global__ __launch_bounds__(256) void gemm_f16_wmma(const _Float16* __restrict__ A,
                                                     const _Float16* __restrict__ Bt,
                                                     float* __restrict__ C,
                                                     int M, int N, int K) {
  const int gwave  = (blockIdx.x * 256 + threadIdx.x) >> 5;
  const int lane   = threadIdx.x & 31;
  const int nstrip = N >> 6;                         // 64-wide strips
  const int strips = (M >> 4) * nstrip;
  if (gwave >= strips) return;                       // wave-uniform
  const int rt = gwave / nstrip;
  const int cs = gwave % nstrip;
  const int ln = lane & 15;
  const int hi = lane >> 4;
  const _Float16* arow = A + (size_t)(rt * 16 + ln) * K;
  const _Float16* br0  = Bt + (size_t)(cs * 64 + 0 * 16 + ln) * K;
  const _Float16* br1  = Bt + (size_t)(cs * 64 + 1 * 16 + ln) * K;
  const _Float16* br2  = Bt + (size_t)(cs * 64 + 2 * 16 + ln) * K;
  const _Float16* br3  = Bt + (size_t)(cs * 64 + 3 * 16 + ln) * K;

  auto loadA = [&](int k0) {
    return cat16(*(const v8h*)(arow + k0 + hi * 8),        // K = hi*8..+7
                 *(const v8h*)(arow + k0 + 16 + hi * 8));  // K = 16+hi*8..+7
  };
  auto loadB = [&](const _Float16* br, int k0) {
    return cat16(*(const v8h*)(br + k0 + hi * 8),          // K = hi*8..+7
                 *(const v8h*)(br + k0 + 16 + hi * 8));    // K = 16+hi*8..+7
  };

  v8f acc0 = {}, acc1 = {}, acc2 = {}, acc3 = {};
  v16h av = loadA(0);
  v16h b0 = loadB(br0, 0), b1 = loadB(br1, 0), b2 = loadB(br2, 0), b3 = loadB(br3, 0);

  int k0 = 0;
  for (; k0 + 32 < K; k0 += 32) {
    const int kn = k0 + 32;
    // issue next step's loads before this step's WMMAs (in-order completion ->
    // partial loadcnt waits; latency hidden under the 4 WMMAs below)
    v16h avn = loadA(kn);
    v16h b0n = loadB(br0, kn);
    v16h b1n = loadB(br1, kn);
    v16h b2n = loadB(br2, kn);
    v16h b3n = loadB(br3, kn);
    if (kn + 32 < K) {                               // prefetch two steps ahead
      __builtin_prefetch(arow + kn + 32, 0, 3);
      __builtin_prefetch(br0 + kn + 32, 0, 3);
      __builtin_prefetch(br1 + kn + 32, 0, 3);
      __builtin_prefetch(br2 + kn + 32, 0, 3);
      __builtin_prefetch(br3 + kn + 32, 0, 3);
    }
    acc0 = __builtin_amdgcn_wmma_f32_16x16x32_f16(false, av, false, b0, (short)0, acc0, false, false);
    acc1 = __builtin_amdgcn_wmma_f32_16x16x32_f16(false, av, false, b1, (short)0, acc1, false, false);
    acc2 = __builtin_amdgcn_wmma_f32_16x16x32_f16(false, av, false, b2, (short)0, acc2, false, false);
    acc3 = __builtin_amdgcn_wmma_f32_16x16x32_f16(false, av, false, b3, (short)0, acc3, false, false);
    av = avn; b0 = b0n; b1 = b1n; b2 = b2n; b3 = b3n;
  }
  // epilogue step
  acc0 = __builtin_amdgcn_wmma_f32_16x16x32_f16(false, av, false, b0, (short)0, acc0, false, false);
  acc1 = __builtin_amdgcn_wmma_f32_16x16x32_f16(false, av, false, b1, (short)0, acc1, false, false);
  acc2 = __builtin_amdgcn_wmma_f32_16x16x32_f16(false, av, false, b2, (short)0, acc2, false, false);
  acc3 = __builtin_amdgcn_wmma_f32_16x16x32_f16(false, av, false, b3, (short)0, acc3, false, false);

  float* crow = C + (size_t)(rt * 16 + hi * 8) * N + cs * 64 + ln;
  #pragma unroll
  for (int r = 0; r < 8; ++r) {
    crow[(size_t)r * N + 0]  = acc0[r];
    crow[(size_t)r * N + 16] = acc1[r];
    crow[(size_t)r * N + 32] = acc2[r];
    crow[(size_t)r * N + 48] = acc3[r];
  }
}

// ---------------------------------------------------------------------------
// Gate combine: h = hf - hb.  Forward rec act = hard_sigmoid, backward = tanh.
// Z layout cols: [i_f, c_f, o_f, i_b, c_b, o_b] x 1024.
// ---------------------------------------------------------------------------
__global__ __launch_bounds__(256) void gates_kernel(const float* __restrict__ Z,
                                                    const float* __restrict__ bf0,
                                                    const float* __restrict__ bb0,
                                                    float* __restrict__ h,
                                                    _Float16* __restrict__ hh) {
  int idx = blockIdx.x * 256 + threadIdx.x;          // 512*1024
  if (idx >= kBW * kE) return;
  int row = idx >> 10, e = idx & (kE - 1);
  const float* zr = Z + (size_t)row * kNG;
  float zi_f = zr[e]            + bf0[e];
  float zc_f = zr[kE + e]       + bf0[2 * kE + e];
  float zo_f = zr[2 * kE + e]   + bf0[3 * kE + e];
  float zi_b = zr[3 * kE + e]   + bb0[e];
  float zc_b = zr[4 * kE + e]   + bb0[2 * kE + e];
  float zo_b = zr[5 * kE + e]   + bb0[3 * kE + e];
  float hf = hsig(zo_f)  * tanhf(hsig(zi_f)  * tanhf(zc_f));
  float hb = tanhf(zo_b) * tanhf(tanhf(zi_b) * tanhf(zc_b));
  float hv = hf - hb;
  h[idx]  = hv;
  hh[idx] = (_Float16)hv;
}

// ---------------------------------------------------------------------------
// Fused attention kernel: one workgroup per (b,w).
//  logits[i,e] = Arow[e] + bd[e] + sum_g H[i,g] * (Wd2s[g,e] + C[g,e])
//  where H[i,g] = h[bw, i*32+g] and C[g,e] = sum_r h[bw, g*32+r]*Wd3[g*32+r, e].
//  Then softmax over e (rows), softmax over i (cols), atomic sum over w.
// D stored TRANSPOSED in LDS (Dt[e][g], 1024x32 f16) so each B fragment is two
// contiguous ds_load_b128 per lane.  A fragments hoisted out of the tile loop;
// next tile's B fragment preloaded before the current WMMA pair.
// LDS: hrow f32[1024] | radd f32[1024] | Dt f16[1024*32] | L f32[32*1025]
// ---------------------------------------------------------------------------
static constexpr int kPitch = kE + 1;                         // bank-conflict-free cols
static constexpr size_t kAttnLds =
    (size_t)kE * 4 + (size_t)kE * 4 + (size_t)kE * 32 * 2 + (size_t)32 * kPitch * 4;

__global__ __launch_bounds__(256) void attn_kernel(const float* __restrict__ h,
                                                   const float* __restrict__ Wd,
                                                   const float* __restrict__ Wd2s,
                                                   const float* __restrict__ Ab,
                                                   const float* __restrict__ bd,
                                                   float* __restrict__ Ssum) {
  const int bw   = blockIdx.x;                       // 0..511
  const int b    = bw >> 5;
  const int tid  = threadIdx.x;                      // 256 threads = 8 waves
  const int wv   = tid >> 5;
  const int lane = tid & 31;
  const int ln   = lane & 15;
  const int hi   = lane >> 4;

  extern __shared__ char smem[];
  float*    hrow = (float*)smem;                     // [1024]
  float*    radd = hrow + kE;                        // [1024]
  _Float16* Dt   = (_Float16*)(radd + kE);           // [1024][32]  (e-major)
  float*    L    = (float*)(Dt + kE * 32);           // [32][kPitch]
  const float* Wd3 = Wd + (size_t)2 * kE * kE;

  for (int e = tid; e < kE; e += 256) {
    hrow[e] = h[(size_t)bw * kE + e];
    radd[e] = Ab[(size_t)bw * kE + e] + bd[e];
  }
  __syncthreads();

  // Dt[e][g] = Wd2s[g][e] + sum_r hrow[32g+r] * Wd3[32g+r][e]
  for (int e = tid; e < kE; e += 256) {
    for (int g = 0; g < 32; ++g) {
      float s = Wd2s[g * kE + e];
      const float* w3 = Wd3 + (size_t)(g * 32) * kE + e;
      const float* hr = hrow + g * 32;
      #pragma unroll 8
      for (int r = 0; r < 32; ++r) s = fmaf(hr[r], w3[(size_t)r * kE], s);
      Dt[e * 32 + g] = (_Float16)s;
    }
  }
  __syncthreads();

  // logits via WMMA: K=32 (one wmma per 16x16 tile).  A fragments (H rows) are
  // loop-invariant -> build once; B fragment for tile nt+8 preloaded before the
  // WMMA pair of tile nt.
  v16h av0 = {}, av1 = {};
  #pragma unroll
  for (int j = 0; j < 8; ++j) {
    av0[j]     = (_Float16)hrow[(0 * 16 + ln) * 32 + hi * 8 + j];       // K=hi*8+j
    av0[8 + j] = (_Float16)hrow[(0 * 16 + ln) * 32 + 16 + hi * 8 + j];  // K=16+hi*8+j
    av1[j]     = (_Float16)hrow[(1 * 16 + ln) * 32 + hi * 8 + j];
    av1[8 + j] = (_Float16)hrow[(1 * 16 + ln) * 32 + 16 + hi * 8 + j];
  }
  auto loadD = [&](int nt) {
    const _Float16* drow = Dt + (size_t)(nt * 16 + ln) * 32 + hi * 8;  // N=ln, K=hi*8.., 16+hi*8..
    return cat16(*(const v8h*)drow, *(const v8h*)(drow + 16));
  };
  v16h bv = loadD(wv);
  for (int nt = wv; nt < 64; nt += 8) {
    v16h bvn = bv;
    if (nt + 8 < 64) bvn = loadD(nt + 8);
    v8f c0 = {}, c1 = {};
    c0 = __builtin_amdgcn_wmma_f32_16x16x32_f16(false, av0, false, bv, (short)0, c0, false, false);
    c1 = __builtin_amdgcn_wmma_f32_16x16x32_f16(false, av1, false, bv, (short)0, c1, false, false);
    #pragma unroll
    for (int r = 0; r < 8; ++r) {
      int e = nt * 16 + ln;
      L[(hi * 8 + r) * kPitch + e]      = c0[r] + radd[e];
      L[(16 + hi * 8 + r) * kPitch + e] = c1[r] + radd[e];
    }
    bv = bvn;
  }
  __syncthreads();

  // softmax over e (last axis) per row i  -- one wave per row
  for (int row = wv; row < 32; row += 8) {
    float mx = -3.4e38f;
    for (int j = 0; j < 32; ++j) mx = fmaxf(mx, L[row * kPitch + lane + 32 * j]);
    for (int off = 16; off; off >>= 1) mx = fmaxf(mx, __shfl_xor(mx, off, 32));
    float s = 0.0f;
    for (int j = 0; j < 32; ++j) {
      float ex = __expf(L[row * kPitch + lane + 32 * j] - mx);
      L[row * kPitch + lane + 32 * j] = ex;
      s += ex;
    }
    for (int off = 16; off; off >>= 1) s += __shfl_xor(s, off, 32);
    float inv = 1.0f / s;
    for (int j = 0; j < 32; ++j) L[row * kPitch + lane + 32 * j] *= inv;
  }
  __syncthreads();

  // softmax over i (axis=2) per column e, then accumulate over w into Ssum[b,i,e]
  for (int e = tid; e < kE; e += 256) {
    float mx = -3.4e38f;
    for (int i = 0; i < 32; ++i) mx = fmaxf(mx, L[i * kPitch + e]);
    float s = 0.0f;
    for (int i = 0; i < 32; ++i) {
      float ex = __expf(L[i * kPitch + e] - mx);
      L[i * kPitch + e] = ex;
      s += ex;
    }
    float inv = 1.0f / s;
    for (int i = 0; i < 32; ++i)
      atomicAdd(&Ssum[((size_t)b * 32 + i) * kE + e], L[i * kPitch + e] * inv);
  }
}

// ---------------------------------------------------------------------------
// Final stage: c = h * Ssum; h2 = [c,h]; two tiny LSTMs (filters=1, tanh);
// softmax over singleton axis -> 1.0 (NaN-propagating).
// One block per (b,w); wave j computes dot j of 8 (dir*4+gate).
// ---------------------------------------------------------------------------
__global__ __launch_bounds__(256) void final_kernel(const float* __restrict__ h,
                                                    const float* __restrict__ Ssum,
                                                    const float* __restrict__ Wf1,
                                                    const float* __restrict__ bf1,
                                                    const float* __restrict__ Wb1,
                                                    const float* __restrict__ bb1,
                                                    float* __restrict__ out) {
  const int bw = blockIdx.x;
  const int b  = bw >> 5;
  const int i  = bw & 31;
  const int tid = threadIdx.x, wv = tid >> 5, lane = tid & 31;
  __shared__ float zsh[8];
  const float* Wm = (wv & 4) ? Wb1 : Wf1;
  const float* bm = (wv & 4) ? bb1 : bf1;
  const int gate = wv & 3;
  float s = 0.0f;
  for (int k = lane; k < 2 * kE; k += 32) {
    float x = (k < kE) ? h[(size_t)bw * kE + k] * Ssum[((size_t)b * 32 + i) * kE + k]
                       : h[(size_t)bw * kE + (k - kE)];
    s = fmaf(x, Wm[(size_t)k * 4 + gate], s);
  }
  for (int off = 16; off; off >>= 1) s += __shfl_xor(s, off, 32);
  if (lane == 0) zsh[wv] = s + bm[gate];
  __syncthreads();
  if (tid == 0) {
    float hf2 = tanhf(zsh[3]) * tanhf(tanhf(zsh[0]) * tanhf(zsh[2]));
    float hb2 = tanhf(zsh[7]) * tanhf(tanhf(zsh[4]) * tanhf(zsh[6]));
    float h2  = hf2 - hb2;
    out[bw] = (h2 - h2) + 1.0f;   // softmax over singleton axis; NaN propagates
  }
}

// ---------------------------------------------------------------------------
// Launcher
// ---------------------------------------------------------------------------
extern "C" void kernel_launch(void* const* d_in, const int* in_sizes, int n_in,
                              void* d_out, int out_size, void* d_ws, size_t ws_size,
                              hipStream_t stream) {
  (void)in_sizes; (void)n_in;
  const float* input_t = (const float*)d_in[0];
  const float* Wf0 = (const float*)d_in[1];
  const float* bf0 = (const float*)d_in[2];
  const float* Wb0 = (const float*)d_in[3];
  const float* bb0 = (const float*)d_in[4];
  // d_in[5]/d_in[6] (ones_a, ones_b) are broadcast-folded analytically.
  const float* Wd  = (const float*)d_in[7];
  const float* bd  = (const float*)d_in[8];
  const float* Wf1 = (const float*)d_in[9];
  const float* bf1 = (const float*)d_in[10];
  const float* Wb1 = (const float*)d_in[11];
  const float* bb1 = (const float*)d_in[12];
  float* out = (float*)d_out;

  // Workspace layout (all sizes power-of-two aligned)
  uint8_t* ws = (uint8_t*)d_ws;
  size_t off = 0;
  auto carve = [&](size_t bytes) { void* p = ws + off; off += (bytes + 255) & ~(size_t)255; return p; };
  _Float16* W1t  = (_Float16*)carve((size_t)kNG * kE * 2);   // 12.0 MiB
  _Float16* Xh   = (_Float16*)carve((size_t)kBW * kE * 2);   //  1.0 MiB
  float*    Z    = (float*)   carve((size_t)kBW * kNG * 4);  // 12.0 MiB
  float*    hbuf = (float*)   carve((size_t)kBW * kE * 4);   //  2.0 MiB
  _Float16* hh   = (_Float16*)carve((size_t)kBW * kE * 2);   //  1.0 MiB
  _Float16* Wd1t = (_Float16*)carve((size_t)kE * kE * 2);    //  2.0 MiB
  float*    Wd2s = (float*)   carve((size_t)32 * kE * 4);    //  128 KiB
  float*    Abuf = (float*)   carve((size_t)kBW * kE * 4);   //  2.0 MiB
  float*    Ssum = (float*)   carve((size_t)kB * kW * kE * 4); // 2.0 MiB

  if (ws_size < off) {  // not enough scratch: output is analytically all-ones anyway
    fill_ones<<<(out_size + 255) / 256, 256, 0, stream>>>(out, out_size);
    return;
  }

  // 1) precision prep
  prep_x_f16<<<(kBW * kE) / 256, 256, 0, stream>>>(input_t, Xh, kBW * kE);
  prep_w1t<<<(kNG * kE) / 256, 256, 0, stream>>>(Wf0, Wb0, W1t);
  prep_wd1t<<<(kE * kE) / 256, 256, 0, stream>>>(Wd, Wd1t);
  prep_wd2s<<<(32 * kE) / 256, 256, 0, stream>>>(Wd, Wd2s);

  // 2) gate GEMM: Z(512x6144) = Xh(512x1024) @ W1t^T   (16x64 strips)
  {
    int strips = (kBW / 16) * (kNG / 64);            // 32*96 = 3072
    gemm_f16_wmma<<<(strips + 7) / 8, 256, 0, stream>>>(Xh, W1t, Z, kBW, kNG, kE);
  }
  // 3) gate activations -> h (f32 + f16)
  gates_kernel<<<(kBW * kE) / 256, 256, 0, stream>>>(Z, bf0, bb0, hbuf, hh);

  // 4) A term: Abuf(512x1024) = h @ Wd1
  {
    int strips = (kBW / 16) * (kE / 64);             // 32*16 = 512
    gemm_f16_wmma<<<(strips + 7) / 8, 256, 0, stream>>>(hh, Wd1t, Abuf, kBW, kE, kE);
  }

  // 5) fused attention (zero the w-accumulator first)
  zero_f32<<<(kB * kW * kE) / 256, 256, 0, stream>>>(Ssum, kB * kW * kE);
  attn_kernel<<<kBW, 256, kAttnLds, stream>>>(hbuf, Wd, Wd2s, Abuf, bd, Ssum);

  // 6) final tiny LSTM pair + singleton softmax
  final_kernel<<<kBW, 256, 0, stream>>>(hbuf, Ssum, Wf1, bf1, Wb1, bb1, out);
}
